// Mamba_4389456576716
// MI455X (gfx1250) — matrix operand order repaired and checked by reading, not compiled, not run on hardware
//
#include <hip/hip_runtime.h>

#define BATCH   16
#define LSEQ    2048
#define CDIM    2048
#define DIN     96
#define DSTATE  16
#define DTRANK  6
#define NROWS   (BATCH * LSEQ)   // 32768 rows of (b,l)
#define NCHUNK  16               // scan chunks per sequence
#define CLEN    (LSEQ / NCHUNK)  // 128 steps per chunk

typedef _Float16 h16;
typedef h16  v8h  __attribute__((ext_vector_type(8)));
typedef h16  v16h __attribute__((ext_vector_type(16)));
typedef float v8f __attribute__((ext_vector_type(8)));

union HV { v16h v; v8h h[2]; };

__device__ __forceinline__ v8f wmma_f16(v16h a, v16h b, v8f c) {
    // D = A(16x32 f16) * B(32x16 f16) + C(16x16 f32)
    return __builtin_amdgcn_wmma_f32_16x16x32_f16(
        /*neg_a=*/false, a, /*neg_b=*/false, b,
        /*c_mod=*/(short)0, c, /*reuse_a=*/false, /*reuse_b=*/false);
}

// ---------------------------------------------------------------------------
// K0: pack weights to f16, K-major so WMMA B-operand loads are contiguous.
//   w1[k*96 + n]   = in_proj_W[n*2048 + k]    (K=2048, N=96)
//   w2[k*2048 + n] = out_proj_W[n*96 + k]     (K=96,   N=2048)
// ---------------------------------------------------------------------------
__global__ void k_pack(const float* __restrict__ inW, const float* __restrict__ outW,
                       h16* __restrict__ w1, h16* __restrict__ w2) {
    int i = blockIdx.x * 256 + threadIdx.x;
    if (i < CDIM * DIN) {
        int k1 = i / DIN, n1 = i % DIN;
        w1[i] = (h16)inW[n1 * CDIM + k1];
        int k2 = i / CDIM, n2 = i % CDIM;
        w2[i] = (h16)outW[n2 * DIN + k2];
    }
}

// ---------------------------------------------------------------------------
// K1: fused LayerNorm + in_proj GEMM.  Block = 16 rows x 256 threads (8 waves).
// ---------------------------------------------------------------------------
#define SA_LD 520   // padded LDS row stride (h16): 1040B -> conflict-free banks

__global__ __launch_bounds__(256)
void k_ln_inproj(const float* __restrict__ x, const float* __restrict__ lnw,
                 const float* __restrict__ lnb, const h16* __restrict__ w1,
                 float* __restrict__ xz) {
    __shared__ __align__(32) h16 s_a[16 * SA_LD];
    __shared__ float s_sum[256], s_sq[256];
    __shared__ float s_mu[16], s_rs[16];

    const int t = threadIdx.x;
    const int rowl = t >> 4, seg = t & 15;
    const size_t rowg = (size_t)blockIdx.x * 16 + rowl;
    const float* xr = x + rowg * CDIM;

    // --- stats (coalesced stride-16 walk) ---
    float sm = 0.f, sq = 0.f;
    for (int i = 0; i < 128; ++i) {
        float v = xr[i * 16 + seg];
        sm += v; sq += v * v;
    }
    s_sum[t] = sm; s_sq[t] = sq;
    __syncthreads();
    if (t < 16) {
        float a = 0.f, b = 0.f;
        for (int j = 0; j < 16; ++j) { a += s_sum[t * 16 + j]; b += s_sq[t * 16 + j]; }
        float mu = a * (1.f / CDIM);
        float var = b * (1.f / CDIM) - mu * mu;
        s_mu[t] = mu;
        s_rs[t] = rsqrtf(var + 1e-5f);
    }
    __syncthreads();

    const int wave = t >> 5, lane = t & 31;
    const int lrow = lane & 15, khalf = (lane >> 4) << 3;
    const float mu = s_mu[rowl], rs = s_rs[rowl];
    v8f acc = {};

    for (int cb = 0; cb < CDIM; cb += 512) {
        // prefetch next chunk of this row while we work on the current one
        if (cb + 512 < CDIM)
            __builtin_prefetch(xr + cb + 512 + seg * 32, 0, 0);
        // stage normalized f16 chunk (16 rows x 512 K), coalesced
        for (int i = 0; i < 32; ++i) {
            int k = cb + i * 16 + seg;
            float v = (xr[k] - mu) * rs * lnw[k] + lnb[k];
            s_a[rowl * SA_LD + (k - cb)] = (h16)v;
        }
        __syncthreads();
        if (wave < 6) {
            const int n0 = wave * 16;
            for (int lk = 0; lk < 512; lk += 32) {
                HV a;
                a.h[0] = *(const v8h*)(s_a + lrow * SA_LD + lk + khalf);
                a.h[1] = *(const v8h*)(s_a + lrow * SA_LD + lk + 16 + khalf);
                v16h b;
                for (int s = 0; s < 16; ++s) {
                    int kg = cb + lk + (s < 8 ? khalf + s : 16 + khalf + (s - 8));
                    b[s] = w1[(size_t)kg * DIN + n0 + lrow];
                }
                acc = wmma_f16(a.v, b, acc);
            }
        }
        __syncthreads();
    }

    if (wave < 6) {
        const int n0 = wave * 16;
        const int col = lane & 15, rbase = (lane >> 4) << 3;
        const size_t rg0 = (size_t)blockIdx.x * 16;
        #pragma unroll
        for (int j = 0; j < 8; ++j)
            xz[(rg0 + rbase + j) * DIN + n0 + col] = acc[j];
    }
}

// ---------------------------------------------------------------------------
// K2: depthwise conv3 (along L, zero pad) + bias + SiLU
// ---------------------------------------------------------------------------
__global__ void k_conv_silu(const float* __restrict__ xz, const float* __restrict__ cw,
                            const float* __restrict__ cb, float* __restrict__ xs) {
    size_t i = (size_t)blockIdx.x * 256 + threadIdx.x;
    if (i >= (size_t)NROWS * DIN) return;
    int c = (int)(i % DIN);
    size_t row = i / DIN;          // b*L + l
    int l = (int)(row % LSEQ);
    float cur  = xz[i];
    float prev = (l > 0)        ? xz[i - DIN] : 0.f;
    float nxt  = (l < LSEQ - 1) ? xz[i + DIN] : 0.f;
    float v = cw[c * 3 + 0] * prev + cw[c * 3 + 1] * cur + cw[c * 3 + 2] * nxt + cb[c];
    xs[i] = v / (1.f + __expf(-v));
}

// ---------------------------------------------------------------------------
// K3: dbc = xs @ x_proj_W.T  (N=38), then delta = softplus(dr@dtW.T+b)*1e-3,
//     split out Bp / Cp.  One small block per row, LDS-broadcast the row.
// ---------------------------------------------------------------------------
__global__ __launch_bounds__(128)
void k_xproj_delta(const float* __restrict__ xs, const float* __restrict__ xW,
                   const float* __restrict__ dtW, const float* __restrict__ dtb,
                   float* __restrict__ delta, float* __restrict__ Bp,
                   float* __restrict__ Cp) {
    __shared__ float s_x[DIN];
    __shared__ float s_dbc[DTRANK + 2 * DSTATE];   // 38
    const size_t row = blockIdx.x;
    const int t = threadIdx.x;
    if (t < DIN) s_x[t] = xs[row * DIN + t];
    __syncthreads();
    if (t < DTRANK + 2 * DSTATE) {
        float a = 0.f;
        const float* w = xW + t * DIN;
        for (int d = 0; d < DIN; ++d) a += s_x[d] * w[d];
        s_dbc[t] = a;
    }
    __syncthreads();
    if (t < DIN) {
        float a = dtb[t];
        #pragma unroll
        for (int j = 0; j < DTRANK; ++j) a += s_dbc[j] * dtW[t * DTRANK + j];
        float sp = (a > 20.f) ? a : log1pf(__expf(a));
        delta[row * DIN + t] = sp * 0.001f;
    }
    if (t < DSTATE) {
        Bp[row * DSTATE + t] = s_dbc[DTRANK + t];
        Cp[row * DSTATE + t] = s_dbc[DTRANK + DSTATE + t];
    }
}

// ---------------------------------------------------------------------------
// K4: chunked (3-pass) selective scan.
//   The recurrence x_t = exp(dl*A)x + dl*B*u is LINEAR, and the chunk product
//   of exp(dl_t*A_n) == exp(A_n * sum(dl_t)), so inter-chunk propagation only
//   needs the per-chunk delta sum — no prefix-product storage.
//   Pass A: 768 blocks, local scan from 0; emit chunk-final state + sum(dl).
//   Pass B: 48 blocks, sequential combine over 16 chunks; emit chunk inits.
//   Pass C: 768 blocks, local scan from true init; emit y.
// ---------------------------------------------------------------------------
__global__ __launch_bounds__(32)
void k_scan_part(const float* __restrict__ xs, const float* __restrict__ delta,
                 const float* __restrict__ Bp, const float* __restrict__ Alog,
                 float* __restrict__ partState, float* __restrict__ dsum) {
    __shared__ float sB[DSTATE];
    const int blk = blockIdx.x;
    const int c = blk % NCHUNK;
    const int dchunk = (blk / NCHUNK) % 3;
    const int b = blk / (NCHUNK * 3);
    const int lane = threadIdx.x;
    const int d = dchunk * 32 + lane;

    float An[DSTATE];
    #pragma unroll
    for (int n = 0; n < DSTATE; ++n) An[n] = -__expf(Alog[d * DSTATE + n]);
    float st[DSTATE];
    #pragma unroll
    for (int n = 0; n < DSTATE; ++n) st[n] = 0.f;
    float S = 0.f;

    const size_t base = (size_t)b * LSEQ + (size_t)c * CLEN;
    for (int l = 0; l < CLEN; ++l) {
        const size_t r = base + l;
        if (lane < 16) sB[lane] = Bp[r * DSTATE + lane];
        __syncthreads();
        const float dl = delta[r * DIN + d];
        const float u  = xs[r * DIN + d];
        S += dl;
        #pragma unroll
        for (int n = 0; n < DSTATE; ++n) {
            float dA = __expf(dl * An[n]);
            st[n] = fmaf(dA, st[n], dl * sB[n] * u);
        }
        __syncthreads();
    }
    const size_t po = ((size_t)(b * NCHUNK + c) * DIN + d) * DSTATE;
    #pragma unroll
    for (int n = 0; n < DSTATE; ++n) partState[po + n] = st[n];
    dsum[(size_t)(b * NCHUNK + c) * DIN + d] = S;
}

__global__ __launch_bounds__(32)
void k_scan_mid(const float* __restrict__ partState, const float* __restrict__ dsum,
                const float* __restrict__ Alog, float* __restrict__ initState) {
    const int b = blockIdx.x / 3;
    const int dchunk = blockIdx.x % 3;
    const int d = dchunk * 32 + threadIdx.x;

    float An[DSTATE];
    #pragma unroll
    for (int n = 0; n < DSTATE; ++n) An[n] = -__expf(Alog[d * DSTATE + n]);
    float X[DSTATE];
    #pragma unroll
    for (int n = 0; n < DSTATE; ++n) X[n] = 0.f;

    for (int c = 0; c < NCHUNK; ++c) {
        const size_t po = ((size_t)(b * NCHUNK + c) * DIN + d) * DSTATE;
        #pragma unroll
        for (int n = 0; n < DSTATE; ++n) initState[po + n] = X[n];   // init of chunk c
        const float S = dsum[(size_t)(b * NCHUNK + c) * DIN + d];
        #pragma unroll
        for (int n = 0; n < DSTATE; ++n) {
            float P = __expf(An[n] * S);          // product of exp(dl*A) over chunk
            X[n] = fmaf(P, X[n], partState[po + n]);
        }
    }
}

__global__ __launch_bounds__(32)
void k_scan_final(const float* __restrict__ xs, const float* __restrict__ delta,
                  const float* __restrict__ Bp, const float* __restrict__ Cp,
                  const float* __restrict__ Alog, const float* __restrict__ Dpar,
                  const float* __restrict__ initState, float* __restrict__ y) {
    __shared__ float sB[DSTATE], sC[DSTATE];
    const int blk = blockIdx.x;
    const int c = blk % NCHUNK;
    const int dchunk = (blk / NCHUNK) % 3;
    const int b = blk / (NCHUNK * 3);
    const int lane = threadIdx.x;
    const int d = dchunk * 32 + lane;

    float An[DSTATE];
    #pragma unroll
    for (int n = 0; n < DSTATE; ++n) An[n] = -__expf(Alog[d * DSTATE + n]);
    const float Dd = Dpar[d];
    float st[DSTATE];
    const size_t po = ((size_t)(b * NCHUNK + c) * DIN + d) * DSTATE;
    #pragma unroll
    for (int n = 0; n < DSTATE; ++n) st[n] = initState[po + n];

    const size_t base = (size_t)b * LSEQ + (size_t)c * CLEN;
    for (int l = 0; l < CLEN; ++l) {
        const size_t r = base + l;
        if (lane < 16) sB[lane] = Bp[r * DSTATE + lane];
        else           sC[lane - 16] = Cp[r * DSTATE + (lane - 16)];
        __syncthreads();
        const float dl = delta[r * DIN + d];
        const float u  = xs[r * DIN + d];
        float yy = 0.f;
        #pragma unroll
        for (int n = 0; n < DSTATE; ++n) {
            float dA = __expf(dl * An[n]);
            st[n] = fmaf(dA, st[n], dl * sB[n] * u);
            yy = fmaf(st[n], sC[n], yy);
        }
        y[r * DIN + d] = fmaf(u, Dd, yy);
        __syncthreads();
    }
}

// ---------------------------------------------------------------------------
// K5: out = y @ out_proj_W.T + residual.  M-tile 16, N-tile 128 (8 waves x 16),
//     K = 96 -> 3 WMMA steps; y-tile staged as f16 in LDS (stride 104).
// ---------------------------------------------------------------------------
#define SY_LD 104

__global__ __launch_bounds__(256)
void k_outproj(const float* __restrict__ y, const h16* __restrict__ w2,
               const float* __restrict__ resid, float* __restrict__ out) {
    __shared__ __align__(32) h16 s_y[16 * SY_LD];
    const int t = threadIdx.x;
    const int mb = blockIdx.x >> 4, nb = blockIdx.x & 15;
    const size_t rg0 = (size_t)mb * 16;

    for (int e = t; e < 16 * DIN; e += 256) {
        int r = e / DIN, k = e % DIN;
        s_y[r * SY_LD + k] = (h16)y[(rg0 + r) * DIN + k];
    }
    __syncthreads();

    const int wave = t >> 5, lane = t & 31;
    const int lrow = lane & 15, khalf = (lane >> 4) << 3;
    const int n0 = nb * 128 + wave * 16;
    v8f acc = {};
    #pragma unroll
    for (int k0 = 0; k0 < DIN; k0 += 32) {
        HV a;
        a.h[0] = *(const v8h*)(s_y + lrow * SY_LD + k0 + khalf);
        a.h[1] = *(const v8h*)(s_y + lrow * SY_LD + k0 + 16 + khalf);
        v16h b;
        for (int s = 0; s < 16; ++s) {
            int kk = k0 + (s < 8 ? khalf + s : 16 + khalf + (s - 8));
            b[s] = w2[(size_t)kk * CDIM + n0 + lrow];
        }
        acc = wmma_f16(a.v, b, acc);
    }
    const int col = lane & 15, rbase = (lane >> 4) << 3;
    #pragma unroll
    for (int j = 0; j < 8; ++j) {
        size_t idx = (rg0 + rbase + j) * CDIM + n0 + col;
        out[idx] = acc[j] + resid[idx];
    }
}

// ---------------------------------------------------------------------------
extern "C" void kernel_launch(void* const* d_in, const int* in_sizes, int n_in,
                              void* d_out, int out_size, void* d_ws, size_t ws_size,
                              hipStream_t stream) {
    (void)in_sizes; (void)n_in; (void)out_size; (void)ws_size;
    const float* feat = (const float*)d_in[0];
    const float* lnw  = (const float*)d_in[1];
    const float* lnb  = (const float*)d_in[2];
    const float* inW  = (const float*)d_in[3];
    const float* cw   = (const float*)d_in[4];
    const float* cb   = (const float*)d_in[5];
    const float* xW   = (const float*)d_in[6];
    const float* dtW  = (const float*)d_in[7];
    const float* dtb  = (const float*)d_in[8];
    const float* Alog = (const float*)d_in[9];
    const float* Dpar = (const float*)d_in[10];
    const float* outW = (const float*)d_in[11];

    char* ws = (char*)d_ws;
    size_t off = 0;
    auto alloc = [&](size_t bytes) {
        size_t r = off;
        off += (bytes + 255) & ~(size_t)255;
        return r;
    };
    h16*   w1   = (h16*)  (ws + alloc((size_t)CDIM * DIN * sizeof(h16)));
    h16*   w2   = (h16*)  (ws + alloc((size_t)DIN * CDIM * sizeof(h16)));
    float* xz   = (float*)(ws + alloc((size_t)NROWS * DIN * sizeof(float)));
    float* xs   = (float*)(ws + alloc((size_t)NROWS * DIN * sizeof(float)));
    float* dl   = (float*)(ws + alloc((size_t)NROWS * DIN * sizeof(float)));
    float* Bp   = (float*)(ws + alloc((size_t)NROWS * DSTATE * sizeof(float)));
    float* Cp   = (float*)(ws + alloc((size_t)NROWS * DSTATE * sizeof(float)));
    float* yv   = (float*)(ws + alloc((size_t)NROWS * DIN * sizeof(float)));
    float* pst  = (float*)(ws + alloc((size_t)BATCH * NCHUNK * DIN * DSTATE * sizeof(float)));
    float* ist  = (float*)(ws + alloc((size_t)BATCH * NCHUNK * DIN * DSTATE * sizeof(float)));
    float* dsum = (float*)(ws + alloc((size_t)BATCH * NCHUNK * DIN * sizeof(float)));

    k_pack<<<(CDIM * DIN + 255) / 256, 256, 0, stream>>>(inW, outW, w1, w2);
    k_ln_inproj<<<NROWS / 16, 256, 0, stream>>>(feat, lnw, lnb, w1, xz);
    k_conv_silu<<<(int)(((size_t)NROWS * DIN + 255) / 256), 256, 0, stream>>>(xz, cw, cb, xs);
    k_xproj_delta<<<NROWS, 128, 0, stream>>>(xs, xW, dtW, dtb, dl, Bp, Cp);
    k_scan_part<<<BATCH * 3 * NCHUNK, 32, 0, stream>>>(xs, dl, Bp, Alog, pst, dsum);
    k_scan_mid<<<BATCH * 3, 32, 0, stream>>>(pst, dsum, Alog, ist);
    k_scan_final<<<BATCH * 3 * NCHUNK, 32, 0, stream>>>(xs, dl, Bp, Cp, Alog, Dpar, ist, yv);
    k_outproj<<<(NROWS / 16) * 16, 256, 0, stream>>>(yv, w2, feat, (float*)d_out);
}
